// SO2Linear_42305427866205
// MI455X (gfx1250) — matrix operand order repaired and checked by reading, not compiled, not run on hardware
//
#include <hip/hip_runtime.h>
#include <hip/hip_bf16.h>

// ---------------------------------------------------------------------------
// SO2Linear for MI455X / gfx1250 (wave32, WMMA).
//
//  out[:,  0: 896] = x[:,  0: 896] @ W0^T + b0                      (K=N=896)
//  out[:,896:2432] = x[:,896:2432] @ Wc1^T   (complex-packed W1)    (K=N=1536)
//  out[:,2432:3712]= x[:,2432:3712]@ Wc2^T   (complex-packed W2)    (K=N=1280)
//
// Weights pre-converted/packed to bf16 in d_ws once per launch. GEMM uses
// v_wmma_f32_16x16x32_bf16, 128x128x32 tiles, 8 wave32s/block, double-
// buffered LDS, branchless steady state, grid N-fastest for L2 A-panel reuse.
// Epilogue has a scalar-uniform full-tile fast path (unguarded stores).
// ---------------------------------------------------------------------------

typedef __attribute__((ext_vector_type(16))) __bf16 v16bf;
typedef __attribute__((ext_vector_type(8)))  float  v8f;

#define BM 128
#define BN 128
#define BK 32
#define LDT 40          // shorts per LDS row: 80B stride (16B aligned, 20-bank skew)
#define NTHREADS 256

union FragU { v16bf v; uint4 q[2]; };

// pack two f32 -> two bf16 (RNE) in one dword (bfe + add3 + perm)
__device__ __forceinline__ unsigned pack2_bf16(float lo, float hi) {
  unsigned a = __builtin_bit_cast(unsigned, lo);
  unsigned b = __builtin_bit_cast(unsigned, hi);
  a = a + 0x7fffu + ((a >> 16) & 1u);              // RNE
  b = b + 0x7fffu + ((b >> 16) & 1u);
  return __builtin_amdgcn_perm(b, a, 0x07060302u); // {b.hi16, a.hi16}
}

__device__ __forceinline__ unsigned short cvt1_bf16(float f) {
  unsigned u = __builtin_bit_cast(unsigned, f);
  u = u + 0x7fffu + ((u >> 16) & 1u);
  return (unsigned short)(u >> 16);
}

__global__ __launch_bounds__(NTHREADS)
void so2_gemm_bf16_wmma(const float* __restrict__ A, int lda,
                        const unsigned short* __restrict__ B, int ldb, // bf16
                        const float* __restrict__ bias,
                        float* __restrict__ C, int ldc,
                        int M, int K)
{
  __shared__ unsigned short As[2][BM][LDT];
  __shared__ unsigned short Bs[2][BN][LDT];

  const int tid   = threadIdx.x;
  const int lane  = tid & 31;
  const int wave  = tid >> 5;
  const int wr    = wave & 1;      // row-group: 64 rows per wave
  const int wc    = wave >> 1;     // col-group: 32 cols per wave
  const int lrow  = lane & 15;
  const int lhalf = lane >> 4;

  const long mBase = (long)blockIdx.y * BM;
  const long nBase = (long)blockIdx.x * BN;   // x fastest -> A panel reused via L2

  // A loader: 128 rows x 32 f32; 4 float4 per thread (rows clamped, not masked)
  const int aRow = tid >> 3;         // 0..31 (+32 per iter)
  const int aCol = (tid & 7) * 4;    // 0,4,...,28
  // B loader: 128 rows x 32 bf16; 1 uint4 (8 bf16) per thread
  const int bRow = tid >> 1;         // 0..127
  const int bCol = (tid & 1) * 16;   // 0 or 16

  const int nChunks = K / BK;

  float4 aReg[4];
  uint4  bReg;

  auto gload = [&](int chunk) {
    const int k0 = chunk * BK;
#pragma unroll
    for (int i = 0; i < 4; ++i) {
      long ar = mBase + aRow + i * 32;
      if (ar > M - 1) ar = M - 1;    // clamp: OOB rows feed never-stored outputs
      aReg[i] = *(const float4*)(A + ar * (long)lda + k0 + aCol);
    }
    bReg = *(const uint4*)(B + (nBase + bRow) * (long)ldb + k0 + bCol);
  };

  auto stlds = [&](int buf) {
#pragma unroll
    for (int i = 0; i < 4; ++i) {
      uint2 ap;
      ap.x = pack2_bf16(aReg[i].x, aReg[i].y);
      ap.y = pack2_bf16(aReg[i].z, aReg[i].w);
      *(uint2*)&As[buf][aRow + i * 32][aCol] = ap;
    }
    *(uint4*)&Bs[buf][bRow][bCol] = bReg;
  };

  v8f acc[4][2];
#pragma unroll
  for (int i = 0; i < 4; ++i)
#pragma unroll
    for (int j = 0; j < 2; ++j)
      acc[i][j] = v8f{0.f, 0.f, 0.f, 0.f, 0.f, 0.f, 0.f, 0.f};

  // A-fragment K offsets per ISA bf16 16x32 layout (two 8-elem runs per lane)
  const int kA0 = lhalf ? 8  : 0;
  const int kA1 = lhalf ? 24 : 16;
  const int kB  = lhalf * 8;         // B: same two 8-elem K runs as A

  auto compute = [&](int buf) {
    v16bf af[4], bf[2];
#pragma unroll
    for (int i = 0; i < 4; ++i) {
      const int row = wr * 64 + i * 16 + lrow;
      FragU u;
      u.q[0] = *(const uint4*)&As[buf][row][kA0];
      u.q[1] = *(const uint4*)&As[buf][row][kA1];
      af[i] = u.v;
    }
#pragma unroll
    for (int j = 0; j < 2; ++j) {
      const int col = wc * 32 + j * 16 + lrow;
      FragU u;
      u.q[0] = *(const uint4*)&Bs[buf][col][kB];
      u.q[1] = *(const uint4*)&Bs[buf][col][kB + 16];
      bf[j] = u.v;
    }
#pragma unroll
    for (int i = 0; i < 4; ++i)
#pragma unroll
      for (int j = 0; j < 2; ++j)
        acc[i][j] = __builtin_amdgcn_wmma_f32_16x16x32_bf16(
            false, af[i], false, bf[j], (short)0, acc[i][j], false, false);
  };

  gload(0);
  stlds(0);
  __syncthreads();

  // branchless steady state; last chunk peeled (nChunks >= 28 for all GEMMs)
  for (int t = 0; t < nChunks - 1; ++t) {
    gload(t + 1);            // prefetch next chunk to regs
    compute(t & 1);          // WMMA on current LDS buffer
    stlds((t + 1) & 1);      // stage into other buffer
    __syncthreads();
  }
  compute((nChunks - 1) & 1);

  // Epilogue: C/D layout -> lanes 0-15: M=v, lanes 16-31: M=v+8, N=lane%16
  const long colBase = nBase + wc * 32 + lrow;
  float bv[2];
#pragma unroll
  for (int j = 0; j < 2; ++j) bv[j] = bias ? bias[colBase + j * 16] : 0.f;

  if (mBase + BM <= (long)M) {
    // full tile: scalar-uniform fast path, unguarded stores
#pragma unroll
    for (int j = 0; j < 2; ++j) {
#pragma unroll
      for (int i = 0; i < 4; ++i) {
        float* Cp = C + (mBase + wr * 64 + i * 16 + lhalf * 8) * (long)ldc
                      + colBase + j * 16;
#pragma unroll
        for (int v = 0; v < 8; ++v)
          Cp[(long)v * ldc] = acc[i][j][v] + bv[j];
      }
    }
  } else {
#pragma unroll
    for (int j = 0; j < 2; ++j) {
#pragma unroll
      for (int i = 0; i < 4; ++i) {
        const long rowBase = mBase + wr * 64 + i * 16 + lhalf * 8;
#pragma unroll
        for (int v = 0; v < 8; ++v) {
          const long r = rowBase + v;
          if (r < M) C[r * (long)ldc + colBase + j * 16] = acc[i][j][v] + bv[j];
        }
      }
    }
  }
}

// Plain f32 -> bf16 weight conversion (for W0).
__global__ void convert_w_bf16(const float* __restrict__ W,
                               unsigned short* __restrict__ Wb, long n)
{
  const long i = (long)blockIdx.x * blockDim.x + threadIdx.x;
  if (i < n) Wb[i] = cvt1_bf16(W[i]);
}

// Build Wc = [[Wr, -Wi], [Wi, Wr]] in bf16 from W = [Wr; Wi] (shape (2*of, din)).
// Turns the SO(2) complex recombination into a single plain GEMM.
__global__ void pack_complex_w_bf16(const float* __restrict__ W,
                                    unsigned short* __restrict__ Wc,
                                    int of, int din)
{
  const long idx   = (long)blockIdx.x * blockDim.x + threadIdx.x;
  const long total = (long)2 * of * 2 * din;
  if (idx >= total) return;
  const int k = (int)(idx % (2 * din));
  const int o = (int)(idx / (2 * din));
  float v;
  if (o < of) {            // pos rows:  [ Wr[o], -Wi[o] ]
    v = (k < din) ? W[(long)o * din + k]
                  : -W[(long)(of + o) * din + (k - din)];
  } else {                 // neg rows:  [ Wi[o'], Wr[o'] ]
    const int oo = o - of;
    v = (k < din) ? W[(long)(of + oo) * din + k]
                  : W[(long)oo * din + (k - din)];
  }
  Wc[idx] = cvt1_bf16(v);
}

extern "C" void kernel_launch(void* const* d_in, const int* in_sizes, int n_in,
                              void* d_out, int out_size, void* d_ws, size_t ws_size,
                              hipStream_t stream) {
  const float* x  = (const float*)d_in[0];   // (E, 29, 128)
  const float* W0 = (const float*)d_in[1];   // (896, 896)
  const float* b0 = (const float*)d_in[2];   // (896,)
  const float* W1 = (const float*)d_in[3];   // (1536, 768)
  const float* W2 = (const float*)d_in[4];   // (1280, 640)

  const int LDX = 29 * 128;                  // 3712
  const int E   = in_sizes[0] / LDX;         // 50000
  float* out = (float*)d_out;

  // bf16 weight staging in workspace (~9.6 MB)
  unsigned short* Wb0 = (unsigned short*)d_ws;           // 896 x 896
  unsigned short* Wb1 = Wb0 + (size_t)896 * 896;         // 1536 x 1536
  unsigned short* Wb2 = Wb1 + (size_t)1536 * 1536;       // 1280 x 1280

  {
    const long n0 = 896L * 896;
    convert_w_bf16<<<dim3((unsigned)((n0 + 255) / 256)), dim3(256), 0, stream>>>(
        W0, Wb0, n0);
    const long t1 = 1536L * 1536;
    pack_complex_w_bf16<<<dim3((unsigned)((t1 + 255) / 256)), dim3(256), 0, stream>>>(
        W1, Wb1, 768, 768);
    const long t2 = 1280L * 1280;
    pack_complex_w_bf16<<<dim3((unsigned)((t2 + 255) / 256)), dim3(256), 0, stream>>>(
        W2, Wb2, 640, 640);
  }

  const unsigned mTiles = (unsigned)((E + BM - 1) / BM);
  dim3 blk(NTHREADS);

  // l=0 block: y0 = x0 @ W0^T + b0
  so2_gemm_bf16_wmma<<<dim3(896 / BN, mTiles), blk, 0, stream>>>(
      x + 0, LDX, Wb0, 896, b0, out + 0, LDX, E, 896);

  // m=1 block: [pos|neg] = [A_r|A_i] @ Wc1^T
  so2_gemm_bf16_wmma<<<dim3(1536 / BN, mTiles), blk, 0, stream>>>(
      x + 896, LDX, Wb1, 1536, nullptr, out + 896, LDX, E, 1536);

  // m=2 block: [pos|neg] = [A_r|A_i] @ Wc2^T
  so2_gemm_bf16_wmma<<<dim3(1280 / BN, mTiles), blk, 0, stream>>>(
      x + 2432, LDX, Wb2, 1280, nullptr, out + 2432, LDX, E, 1280);
}
